// AttentionBlock_71365176590699
// MI455X (gfx1250) — matrix operand order repaired and checked by reading, not compiled, not run on hardware
//
#include <hip/hip_runtime.h>

typedef __bf16 bf16;
typedef __attribute__((ext_vector_type(16))) __bf16   v16bf;
typedef __attribute__((ext_vector_type(8)))  float    v8f;
typedef __attribute__((ext_vector_type(4)))  unsigned u32x4;

union ABfrag { v16bf v; u32x4 q[2]; };
union Pack8  { u32x4 q; unsigned short h[8]; };

__device__ __forceinline__ unsigned short f2bf_bits(float f) {
  unsigned u = __builtin_bit_cast(unsigned, f);
  return (unsigned short)((u + 0x7FFFu + ((u >> 16) & 1u)) >> 16);  // RNE
}
__device__ __forceinline__ bf16 f2bf(float f) {
  unsigned short s = f2bf_bits(f);
  return __builtin_bit_cast(bf16, s);
}
__device__ __forceinline__ v8f vzero8() {
  v8f r;
#pragma unroll
  for (int i = 0; i < 8; ++i) r[i] = 0.0f;
  return r;
}
__device__ __forceinline__ v8f wmma_bf16(v16bf a, v16bf b, v8f c) {
  // D(f32 16x16) = A(bf16 16x32) x B(bf16 32x16) + C
  return __builtin_amdgcn_wmma_f32_16x16x32_bf16(false, a, false, b, (short)0, c,
                                                 false, false);
}

// Async direct-to-LDS 16B copy (CDNA5 VGLOBAL GVS form, ASYNCcnt-tracked):
//   global_load_async_to_lds_b128 vdst(LDS byte offset), vaddr(i32 offset), saddr
__device__ __forceinline__ void async_g2l_b128(unsigned lds_off, unsigned goff,
                                               const bf16* sbase) {
  asm volatile("global_load_async_to_lds_b128 %0, %1, %2"
               :: "v"(lds_off), "v"(goff), "s"(sbase) : "memory");
}
__device__ __forceinline__ void wait_async_le8() {   // prev tile done, next in flight
  asm volatile("s_wait_asynccnt 8" ::: "memory");
}
__device__ __forceinline__ void wait_async_le0() {
  asm volatile("s_wait_asynccnt 0" ::: "memory");
}

// B=16, C=512, T=1024, G=32 (16 ch/group), heads=8, ch=64.

// ---------------- Kernel 1: weights f32 -> bf16 ----------------
__global__ __launch_bounds__(256) void k_convert_w(
    const float* __restrict__ wqkv, const float* __restrict__ wproj,
    bf16* __restrict__ owqkv, bf16* __restrict__ owproj) {
  int i = blockIdx.x * 256 + threadIdx.x;
  if (i < 1536 * 512) owqkv[i] = f2bf(wqkv[i]);
  if (i < 512 * 512)  owproj[i] = f2bf(wproj[i]);
}

// ---------------- Kernel 2: GroupNorm -> xnT[b][t][c] bf16 ----------------
__global__ __launch_bounds__(256) void k_groupnorm(
    const float* __restrict__ x, const float* __restrict__ gamma,
    const float* __restrict__ beta, bf16* __restrict__ xnT) {
  const int b = blockIdx.x, g = blockIdx.y;
  const int tid = threadIdx.x;
  const float* xb = x + ((size_t)b * 512 + (size_t)g * 16) * 1024;  // 16 ch x 1024 t
  float s = 0.f, s2 = 0.f;
  for (int i = tid; i < 16384; i += 256) {
    float v = xb[i];
    s += v; s2 += v * v;
  }
  __shared__ float r1[256], r2[256];
  r1[tid] = s; r2[tid] = s2;
  __syncthreads();
  for (int off = 128; off > 0; off >>= 1) {
    if (tid < off) { r1[tid] += r1[tid + off]; r2[tid] += r2[tid + off]; }
    __syncthreads();
  }
  const float mean = r1[0] * (1.0f / 16384.0f);
  const float var  = r2[0] * (1.0f / 16384.0f) - mean * mean;
  const float inv  = rsqrtf(var + 1e-5f);
  for (int i = tid; i < 16384; i += 256) {
    int cc = i & 15, t = i >> 4;
    int c = g * 16 + cc;
    float v = (xb[cc * 1024 + t] - mean) * inv;
    v = v * gamma[c] + beta[c];
    xnT[((size_t)b * 1024 + t) * 512 + c] = f2bf(v);
  }
}

// ---------------- Kernel 3: QKV GEMM (per batch, M=1536 N=1024 K=512) ------
// A = w_qkv[o][c] (row-major, K contiguous), B from xnT[b][t][c] (K contiguous).
// Block = 4 waves, 128x128 tile; wave = 64x64 (4x4 WMMA accumulators).
// Double-buffered async direct-to-LDS staging (ASYNCcnt): tile i+1 streams in
// while WMMAs consume tile i.
__global__ __launch_bounds__(128) void k_qkv_gemm(
    const bf16* __restrict__ wq, const float* __restrict__ bqkv,
    const bf16* __restrict__ xnT, bf16* __restrict__ qT,
    bf16* __restrict__ kT, bf16* __restrict__ vm) {
  const int b = blockIdx.x, mblk = blockIdx.y, nblk = blockIdx.z;
  const int tid = threadIdx.x;
  const int lane = tid & 31, wave = tid >> 5;
  const int wm = wave >> 1, wn = wave & 1;
  const int lh = lane >> 4, l16 = lane & 15;

  __shared__ __attribute__((aligned(16))) bf16 lA[2][128 * 32];  // [buf][m][k]
  __shared__ __attribute__((aligned(16))) bf16 lB[2][128 * 32];  // [buf][n(t)][k(c)]

  v8f acc[4][4];
#pragma unroll
  for (int mi = 0; mi < 4; ++mi)
#pragma unroll
    for (int ni = 0; ni < 4; ++ni) acc[mi][ni] = vzero8();

  const bf16* Ab = wq + (size_t)(mblk * 128) * 512;
  const bf16* Bb = xnT + ((size_t)b * 1024 + (size_t)nblk * 128) * 512;
  const unsigned ldsA0 = (unsigned)(size_t)(&lA[0][tid * 32]);
  const unsigned ldsA1 = (unsigned)(size_t)(&lA[1][tid * 32]);
  const unsigned ldsB0 = (unsigned)(size_t)(&lB[0][tid * 32]);
  const unsigned ldsB1 = (unsigned)(size_t)(&lB[1][tid * 32]);

  // prologue: stage tile 0 into buffer 0 (one 64B row per thread, A and B)
  {
    const unsigned goff = (unsigned)(tid * 1024);
#pragma unroll
    for (int u = 0; u < 4; ++u) {
      async_g2l_b128(ldsA0 + u * 16, goff + u * 16, Ab);
      async_g2l_b128(ldsB0 + u * 16, goff + u * 16, Bb);
    }
  }

  for (int it = 0; it < 16; ++it) {
    if (it + 1 < 16) {
      const unsigned nA = (it & 1) ? ldsA0 : ldsA1;
      const unsigned nB = (it & 1) ? ldsB0 : ldsB1;
      const unsigned goff = (unsigned)(tid * 1024 + (it + 1) * 64);
#pragma unroll
      for (int u = 0; u < 4; ++u) {
        async_g2l_b128(nA + u * 16, goff + u * 16, Ab);
        async_g2l_b128(nB + u * 16, goff + u * 16, Bb);
      }
      if (it + 2 < 16) {
        __builtin_prefetch(Ab + (size_t)tid * 512 + (it + 2) * 32, 0, 3);
        __builtin_prefetch(Bb + (size_t)tid * 512 + (it + 2) * 32, 0, 3);
      }
      wait_async_le8();  // tile `it` landed (async loads complete in order)
    } else {
      wait_async_le0();
    }
    __syncthreads();
    const bf16* curA = lA[it & 1];
    const bf16* curB = lB[it & 1];
    ABfrag af[4], bfr[4];
#pragma unroll
    for (int mi = 0; mi < 4; ++mi) {
      int row = wm * 64 + mi * 16 + l16;
      af[mi].q[0] = *reinterpret_cast<const u32x4*>(&curA[row * 32 + lh * 8]);
      af[mi].q[1] = *reinterpret_cast<const u32x4*>(&curA[row * 32 + 16 + lh * 8]);
    }
#pragma unroll
    for (int ni = 0; ni < 4; ++ni) {
      int row = wn * 64 + ni * 16 + l16;
      bfr[ni].q[0] = *reinterpret_cast<const u32x4*>(&curB[row * 32 + lh * 8]);
      bfr[ni].q[1] = *reinterpret_cast<const u32x4*>(&curB[row * 32 + 16 + lh * 8]);
    }
#pragma unroll
    for (int mi = 0; mi < 4; ++mi)
#pragma unroll
      for (int ni = 0; ni < 4; ++ni)
        acc[mi][ni] = wmma_bf16(af[mi].v, bfr[ni].v, acc[mi][ni]);
    __syncthreads();  // all waves done with this buffer before it is re-staged
  }

  // Epilogue: o<512 -> q, <1024 -> k, else v. Fold bias; scale q,k by 64^-0.25.
  const int oblock = mblk * 128 + wm * 64;
  const int region = oblock >> 9;
  const float qs = 0.35355339059327373f;
#pragma unroll
  for (int mi = 0; mi < 4; ++mi) {
    const int o0 = oblock + mi * 16 + lh * 8;  // first of this lane's 8 rows
    const int oo = o0 & 511;
    const int h = oo >> 6;
    const int chb = oo & 63;
    const int bh = b * 8 + h;
    float bias[8];
#pragma unroll
    for (int r = 0; r < 8; ++r) bias[r] = bqkv[o0 + r];
#pragma unroll
    for (int ni = 0; ni < 4; ++ni) {
      const int t = nblk * 128 + wn * 64 + ni * 16 + l16;
      if (region == 0) {
        Pack8 p;
#pragma unroll
        for (int r = 0; r < 8; ++r) p.h[r] = f2bf_bits((acc[mi][ni][r] + bias[r]) * qs);
        *reinterpret_cast<u32x4*>(&qT[((size_t)bh * 1024 + t) * 64 + chb]) = p.q;
      } else if (region == 1) {
        Pack8 p;
#pragma unroll
        for (int r = 0; r < 8; ++r) p.h[r] = f2bf_bits((acc[mi][ni][r] + bias[r]) * qs);
        *reinterpret_cast<u32x4*>(&kT[((size_t)bh * 1024 + t) * 64 + chb]) = p.q;
      } else {
#pragma unroll
        for (int r = 0; r < 8; ++r)
          vm[((size_t)bh * 64 + chb + r) * 1024 + t] = f2bf(acc[mi][ni][r] + bias[r]);
      }
    }
  }
}

// ---------------- Kernel 4: flash attention, 1 wave per 16-query tile ------
__global__ __launch_bounds__(128) void k_attn(
    const bf16* __restrict__ qT, const bf16* __restrict__ kT,
    const bf16* __restrict__ vm, bf16* __restrict__ aT) {
  const int bh = blockIdx.x;
  const int b = bh >> 3, h = bh & 7;
  const int wave = threadIdx.x >> 5, lane = threadIdx.x & 31;
  const int lh = lane >> 4, l16 = lane & 15;
  const int t0 = (blockIdx.y * 4 + wave) * 16;

  __shared__ __attribute__((aligned(16))) bf16 lP[4][16 * 32];  // per-wave P tile
  bf16* Pw = lP[wave];

  // Q A-fragments (K=64 -> two 16x32 A tiles)
  const bf16* qrow = qT + ((size_t)bh * 1024 + t0 + l16) * 64;
  ABfrag Aq0, Aq1;
  Aq0.q[0] = *reinterpret_cast<const u32x4*>(qrow + lh * 8);
  Aq0.q[1] = *reinterpret_cast<const u32x4*>(qrow + 16 + lh * 8);
  Aq1.q[0] = *reinterpret_cast<const u32x4*>(qrow + 32 + lh * 8);
  Aq1.q[1] = *reinterpret_cast<const u32x4*>(qrow + 48 + lh * 8);

  float mrun[8], lrun[8];
#pragma unroll
  for (int r = 0; r < 8; ++r) { mrun[r] = -1e30f; lrun[r] = 0.f; }
  v8f acc[4];
#pragma unroll
  for (int j = 0; j < 4; ++j) acc[j] = vzero8();

  for (int s0 = 0; s0 < 1024; s0 += 32) {
    v8f lg[2];
#pragma unroll
    for (int st = 0; st < 2; ++st) {
      const bf16* krow = kT + ((size_t)bh * 1024 + s0 + st * 16 + l16) * 64;
      ABfrag B0, B1;
      B0.q[0] = *reinterpret_cast<const u32x4*>(krow + lh * 8);
      B0.q[1] = *reinterpret_cast<const u32x4*>(krow + 16 + lh * 8);
      B1.q[0] = *reinterpret_cast<const u32x4*>(krow + 32 + lh * 8);
      B1.q[1] = *reinterpret_cast<const u32x4*>(krow + 48 + lh * 8);
      v8f t = vzero8();
      t = wmma_bf16(Aq0.v, B0.v, t);
      t = wmma_bf16(Aq1.v, B1.v, t);
      lg[st] = t;
    }
    // online softmax over the 32 new columns (row lives in half-wave lanes)
#pragma unroll
    for (int r = 0; r < 8; ++r) {
      float v0 = lg[0][r], v1 = lg[1][r];
      float mx = fmaxf(v0, v1);
      mx = fmaxf(mx, __shfl_xor(mx, 1));
      mx = fmaxf(mx, __shfl_xor(mx, 2));
      mx = fmaxf(mx, __shfl_xor(mx, 4));
      mx = fmaxf(mx, __shfl_xor(mx, 8));
      float mnew = fmaxf(mrun[r], mx);
      float corr = __expf(mrun[r] - mnew);
      float p0 = __expf(v0 - mnew);
      float p1 = __expf(v1 - mnew);
      float rs = p0 + p1;
      rs += __shfl_xor(rs, 1);
      rs += __shfl_xor(rs, 2);
      rs += __shfl_xor(rs, 4);
      rs += __shfl_xor(rs, 8);
      lrun[r] = lrun[r] * corr + rs;
      mrun[r] = mnew;
#pragma unroll
      for (int j = 0; j < 4; ++j) acc[j][r] *= corr;
      const int row = lh * 8 + r;  // C-layout row -> P row
      Pw[row * 32 + l16]      = f2bf(p0);
      Pw[row * 32 + 16 + l16] = f2bf(p1);
    }
    asm volatile("s_wait_dscnt 0" ::: "memory");  // P tile visible wave-wide
    ABfrag Ap;
    Ap.q[0] = *reinterpret_cast<const u32x4*>(&Pw[l16 * 32 + lh * 16]);
    Ap.q[1] = *reinterpret_cast<const u32x4*>(&Pw[l16 * 32 + lh * 16 + 8]);
#pragma unroll
    for (int j = 0; j < 4; ++j) {
      const bf16* vrow = vm + ((size_t)bh * 64 + j * 16 + l16) * 1024 + s0 + lh * 16;
      ABfrag Bv;
      Bv.q[0] = *reinterpret_cast<const u32x4*>(vrow);
      Bv.q[1] = *reinterpret_cast<const u32x4*>(vrow + 8);
      acc[j] = wmma_bf16(Ap.v, Bv.v, acc[j]);
    }
    if (s0 + 32 < 1024)
      __builtin_prefetch(kT + ((size_t)bh * 1024 + s0 + 32 + l16) * 64, 0, 3);
  }

  // normalize and write aT[b][t][c], c = h*64 + j*16 + l16
#pragma unroll
  for (int r = 0; r < 8; ++r) {
    const float inv = 1.0f / lrun[r];
    const int t = t0 + lh * 8 + r;
    bf16* arow = aT + ((size_t)b * 1024 + t) * 512 + h * 64;
#pragma unroll
    for (int j = 0; j < 4; ++j) arow[j * 16 + l16] = f2bf(acc[j][r] * inv);
  }
}

// ---------------- Kernel 5: proj GEMM + bias + residual --------------------
__global__ __launch_bounds__(128) void k_proj_gemm(
    const bf16* __restrict__ wp, const float* __restrict__ bproj,
    const bf16* __restrict__ aT, const float* __restrict__ x,
    float* __restrict__ out) {
  const int b = blockIdx.x, mblk = blockIdx.y, nblk = blockIdx.z;
  const int tid = threadIdx.x;
  const int lane = tid & 31, wave = tid >> 5;
  const int wm = wave >> 1, wn = wave & 1;
  const int lh = lane >> 4, l16 = lane & 15;

  __shared__ __attribute__((aligned(16))) bf16 lA[2][128 * 32];
  __shared__ __attribute__((aligned(16))) bf16 lB[2][128 * 32];

  v8f acc[4][4];
#pragma unroll
  for (int mi = 0; mi < 4; ++mi)
#pragma unroll
    for (int ni = 0; ni < 4; ++ni) acc[mi][ni] = vzero8();

  const bf16* Ab = wp + (size_t)(mblk * 128) * 512;
  const bf16* Bb = aT + ((size_t)b * 1024 + (size_t)nblk * 128) * 512;
  const unsigned ldsA0 = (unsigned)(size_t)(&lA[0][tid * 32]);
  const unsigned ldsA1 = (unsigned)(size_t)(&lA[1][tid * 32]);
  const unsigned ldsB0 = (unsigned)(size_t)(&lB[0][tid * 32]);
  const unsigned ldsB1 = (unsigned)(size_t)(&lB[1][tid * 32]);

  {
    const unsigned goff = (unsigned)(tid * 1024);
#pragma unroll
    for (int u = 0; u < 4; ++u) {
      async_g2l_b128(ldsA0 + u * 16, goff + u * 16, Ab);
      async_g2l_b128(ldsB0 + u * 16, goff + u * 16, Bb);
    }
  }

  for (int it = 0; it < 16; ++it) {
    if (it + 1 < 16) {
      const unsigned nA = (it & 1) ? ldsA0 : ldsA1;
      const unsigned nB = (it & 1) ? ldsB0 : ldsB1;
      const unsigned goff = (unsigned)(tid * 1024 + (it + 1) * 64);
#pragma unroll
      for (int u = 0; u < 4; ++u) {
        async_g2l_b128(nA + u * 16, goff + u * 16, Ab);
        async_g2l_b128(nB + u * 16, goff + u * 16, Bb);
      }
      if (it + 2 < 16) {
        __builtin_prefetch(Ab + (size_t)tid * 512 + (it + 2) * 32, 0, 3);
        __builtin_prefetch(Bb + (size_t)tid * 512 + (it + 2) * 32, 0, 3);
      }
      wait_async_le8();
    } else {
      wait_async_le0();
    }
    __syncthreads();
    const bf16* curA = lA[it & 1];
    const bf16* curB = lB[it & 1];
    ABfrag af[4], bfr[4];
#pragma unroll
    for (int mi = 0; mi < 4; ++mi) {
      int row = wm * 64 + mi * 16 + l16;
      af[mi].q[0] = *reinterpret_cast<const u32x4*>(&curA[row * 32 + lh * 8]);
      af[mi].q[1] = *reinterpret_cast<const u32x4*>(&curA[row * 32 + 16 + lh * 8]);
    }
#pragma unroll
    for (int ni = 0; ni < 4; ++ni) {
      int row = wn * 64 + ni * 16 + l16;
      bfr[ni].q[0] = *reinterpret_cast<const u32x4*>(&curB[row * 32 + lh * 8]);
      bfr[ni].q[1] = *reinterpret_cast<const u32x4*>(&curB[row * 32 + 16 + lh * 8]);
    }
#pragma unroll
    for (int mi = 0; mi < 4; ++mi)
#pragma unroll
      for (int ni = 0; ni < 4; ++ni)
        acc[mi][ni] = wmma_bf16(af[mi].v, bfr[ni].v, acc[mi][ni]);
    __syncthreads();
  }

#pragma unroll
  for (int mi = 0; mi < 4; ++mi) {
    const int o0 = mblk * 128 + wm * 64 + mi * 16 + lh * 8;
    float bias[8];
#pragma unroll
    for (int r = 0; r < 8; ++r) bias[r] = bproj[o0 + r];
#pragma unroll
    for (int ni = 0; ni < 4; ++ni) {
      const int t = nblk * 128 + wn * 64 + ni * 16 + l16;
#pragma unroll
      for (int r = 0; r < 8; ++r) {
        size_t idx = ((size_t)b * 512 + o0 + r) * 1024 + t;
        out[idx] = x[idx] + acc[mi][ni][r] + bias[r];
      }
    }
  }
}

// ---------------- Host launcher ----------------
extern "C" void kernel_launch(void* const* d_in, const int* in_sizes, int n_in,
                              void* d_out, int out_size, void* d_ws, size_t ws_size,
                              hipStream_t stream) {
  (void)in_sizes; (void)n_in; (void)out_size; (void)ws_size;
  const float* x     = (const float*)d_in[0];
  const float* gamma = (const float*)d_in[1];
  const float* beta  = (const float*)d_in[2];
  const float* wqkv  = (const float*)d_in[3];
  const float* bqkv  = (const float*)d_in[4];
  const float* wproj = (const float*)d_in[5];
  const float* bproj = (const float*)d_in[6];
  float* out = (float*)d_out;

  char* ws = (char*)d_ws;
  size_t off = 0;
  bf16* wq_bf = (bf16*)(ws + off); off += (size_t)1536 * 512 * 2;
  bf16* wp_bf = (bf16*)(ws + off); off += (size_t)512 * 512 * 2;
  bf16* xnT   = (bf16*)(ws + off); off += (size_t)16 * 1024 * 512 * 2;
  bf16* qT    = (bf16*)(ws + off); off += (size_t)128 * 1024 * 64 * 2;
  bf16* kT    = (bf16*)(ws + off); off += (size_t)128 * 1024 * 64 * 2;
  bf16* vm    = (bf16*)(ws + off); off += (size_t)128 * 64 * 1024 * 2;
  bf16* aT    = (bf16*)(ws + off); off += (size_t)16 * 1024 * 512 * 2;
  (void)off;

  k_convert_w<<<3072, 256, 0, stream>>>(wqkv, wproj, wq_bf, wp_bf);
  k_groupnorm<<<dim3(16, 32), 256, 0, stream>>>(x, gamma, beta, xnT);
  k_qkv_gemm<<<dim3(16, 12, 8), 128, 0, stream>>>(wq_bf, bqkv, xnT, qT, kT, vm);
  k_attn<<<dim3(128, 16), 128, 0, stream>>>(qT, kT, vm, aT);
  k_proj_gemm<<<dim3(16, 4, 8), 128, 0, stream>>>(wp_bf, bproj, aT, x, out);
}
